// HL_HGCNN_ABCD_dense_int3_attpool_87247965651034
// MI455X (gfx1250) — matrix operand order repaired and checked by reading, not compiled, not run on hardware
//
#include <hip/hip_runtime.h>
#include <hip/hip_bf16.h>
#include <cstdint>
#include <cstddef>

typedef __attribute__((ext_vector_type(16))) _Float16 v16h;
typedef __attribute__((ext_vector_type(8)))  _Float16 v8h;
typedef __attribute__((ext_vector_type(8)))  float    v8f;

// ---------------- problem constants (from reference) ----------------
static constexpr int NT1 = 131072, NS1 = 262144, NT2 = 16000, NS2 = 17424;
static constexpr int MT1 = 786432, MS1 = 1048576, MT2 = 96000, MS2 = 104544;
static constexpr int FT = 74, FS = 11;
static constexpr float LEAK = 0.1f;

static constexpr int NLAY = 6;
static constexpr int Fl[NLAY] = {64, 64, 128, 128, 256, 256};   // output width per layer
static constexpr int Dl[NLAY] = {64, 128, 192, 320, 448, 704};  // xt0/xs0 width before layer

// params leaf indices (jax pytree sorted-key flatten of the params dict)
enum {
  P_att_Ws = 0, P_att_Wt = 1, P_att_bs = 2, P_att_bt = 3,
  P_cv_base = 4,           // +block*8 : sW,sb,sbe,sg,tW,tb,tbe,tg
  P_emb_W = 52, P_emb_b = 53,
  P_init_s_W = 54, P_init_s_b = 55, P_init_s_be = 56, P_init_s_g = 57,
  P_init_t_W = 58, P_init_t_b = 59, P_init_t_be = 60, P_init_t_g = 61,
  P_ni_base = 62,          // +block*4 : Ws,Wt,bs,bt
  P_out_W = 86, P_out_b = 87,
  P_ro_sW = 88, P_ro_sb = 89, P_ro_tW = 90, P_ro_tb = 91
};

// ---------------- small device helpers ----------------
__device__ __forceinline__ long gidx() { return (long)blockIdx.x * blockDim.x + threadIdx.x; }
__device__ __forceinline__ long gstr() { return (long)gridDim.x * blockDim.x; }

// ---------------- WMMA GEMM: C[M,N] = A[M,K] * Bt[N,K]^T (+bias | +=C) ----------------
// A: f16 row-major, lda = Kp (multiple of 32, zero padded)
// Bt: f16, weight stored transposed (N rows of K), ldb = Kp
// C: f32, ldc = Nact (masked stores for padded N tiles)
__global__ void __launch_bounds__(256)
k_wmma_gemm(const _Float16* __restrict__ A, const _Float16* __restrict__ Bt,
            const float* __restrict__ bias, float* __restrict__ C,
            int lda, int ldb, int ldc, int Mtiles, int Ntiles, int Nact, int K,
            int accumulate, int leaky)
{
  int wave = blockIdx.x * (blockDim.x >> 5) + (threadIdx.x >> 5);
  int total = Mtiles * Ntiles;
  if (wave >= total) return;               // whole-wave uniform exit: EXEC all-ones below
  int tn = wave % Ntiles, tm = wave / Ntiles;
  int lane = threadIdx.x & 31;
  int lr = lane & 15;
  int hi = lane >> 4;                      // lane group 0/1
  int col = tn * 16 + lr;                  // C column owned by this lane
  const _Float16* Arow = A  + (size_t)(tm * 16 + lr) * lda;
  const _Float16* Brow = Bt + (size_t)col * ldb;

  v8f c;
  if (accumulate) {
#pragma unroll
    for (int r = 0; r < 8; ++r) {
      int m = tm * 16 + r + hi * 8;
      c[r] = (col < Nact) ? C[(size_t)m * ldc + col] : 0.f;
    }
  } else {
    float b = 0.f;
    if (bias != nullptr && col < Nact) b = bias[col];
#pragma unroll
    for (int r = 0; r < 8; ++r) c[r] = b;
  }

  for (int k0 = 0; k0 < K; k0 += 32) {
    // A fragment (contiguous K, matching B):
    //   lanes 0-15: K = k0+0..15 ; lanes 16-31: K = k0+16..31
    v8h a0 = *(const v8h*)(Arow + k0 + hi * 16);
    v8h a1 = *(const v8h*)(Arow + k0 + hi * 16 + 8);
    // B fragment: lanes 0-15 hold K = k0+0..15 (contiguous), lanes 16-31 K = k0+16..31
    v16h b = *(const v16h*)(Brow + k0 + hi * 16);
    v16h a;
#pragma unroll
    for (int i = 0; i < 8; ++i) { a[i] = a0[i]; a[8 + i] = a1[i]; }
    c = __builtin_amdgcn_wmma_f32_16x16x32_f16(false, a, false, b, (short)0, c, false, false);
  }

  if (col < Nact) {
#pragma unroll
    for (int r = 0; r < 8; ++r) {
      int m = tm * 16 + r + hi * 8;
      float v = c[r];
      if (leaky) v = v > 0.f ? v : LEAK * v;
      C[(size_t)m * ldc + col] = v;
    }
  }
}

// ---------------- elementwise / sparse kernels ----------------
__global__ void k_wT_f16(const float* __restrict__ W, int K, int N,
                         _Float16* __restrict__ dst, int Kp, int Np, long total) {
  for (long i = gidx(); i < total; i += gstr()) {
    int n = (int)(i / Kp), k = (int)(i % Kp);
    dst[i] = (k < K && n < N) ? (_Float16)W[(size_t)k * N + n] : (_Float16)0.f;
  }
}

__global__ void k_tof16(const float* __restrict__ X, int px, int coff, int F,
                        _Float16* __restrict__ A, int Kp, long total) {
  for (long i = gidx(); i < total; i += gstr()) {
    long r = i / Kp; int k = (int)(i % Kp);
    A[i] = (k < F) ? (_Float16)X[r * (long)px + coff + k] : (_Float16)0.f;
  }
}

__global__ void k_spmm(const int* __restrict__ ei, const float* __restrict__ ew, int E,
                       const float* __restrict__ X, int px, int F,
                       float* __restrict__ Y, long total) {
  for (long i = gidx(); i < total; i += gstr()) {
    long e = i / F; int c = (int)(i % F);
    atomicAdd(&Y[(long)ei[e] * F + c], ew[e] * X[(long)ei[E + e] * px + c]);
  }
}

__global__ void k_xmLx(const float* __restrict__ X, int px, int F,
                       float* __restrict__ T, long total) {
  for (long i = gidx(); i < total; i += gstr()) {
    long r = i / F; int c = (int)(i % F);
    T[i] = X[r * (long)px + c] - T[i];
  }
}

__global__ void k_colstats(const float* __restrict__ X, int M, int F, int rpb,
                           float* __restrict__ s, float* __restrict__ q) {
  int c = threadIdx.x;                      // blockDim.x == F (64/128/256)
  long r0 = (long)blockIdx.x * rpb, r1 = r0 + rpb; if (r1 > M) r1 = M;
  float a = 0.f, b = 0.f;
  for (long r = r0; r < r1; ++r) { float v = X[r * F + c]; a += v; b += v * v; }
  atomicAdd(&s[c], a); atomicAdd(&q[c], b);
}

__global__ void k_bn_leaky(const float* __restrict__ X, int M, int F,
                           const float* __restrict__ s, const float* __restrict__ q,
                           const float* __restrict__ g, const float* __restrict__ be,
                           float* __restrict__ OUT, int pout, int coff, long total) {
  float invM = 1.f / (float)M;
  for (long i = gidx(); i < total; i += gstr()) {
    long r = i / F; int c = (int)(i % F);
    float m = s[c] * invM;
    float v = q[c] * invM - m * m;
    float y = (X[i] - m) * rsqrtf(v + 1e-5f) * g[c] + be[c];
    OUT[r * (long)pout + coff + c] = y > 0.f ? y : LEAK * y;
  }
}

__global__ void k_scatter_mt(const int* __restrict__ incp, int NS,
                             const float* __restrict__ XS, int ps, int coff, int d,
                             float* __restrict__ MT, long total) {
  for (long i = gidx(); i < total; i += gstr()) {
    long e = i / d; int c = (int)(i % d);
    float v = XS[e * (long)ps + coff + c];
    atomicAdd(&MT[(long)incp[e] * d + c], -v);        // src
    atomicAdd(&MT[(long)incp[NS + e] * d + c], v);    // dst
  }
}

__global__ void k_concat_mt_f16(const float* __restrict__ XT, int pt,
                                const float* __restrict__ MT, int d,
                                const float* __restrict__ Dinv,
                                _Float16* __restrict__ A, int Kp, long total) {
  for (long i = gidx(); i < total; i += gstr()) {
    long r = i / Kp; int k = (int)(i % Kp);
    float v = (k < d) ? XT[r * (long)pt + k] : MT[r * (long)d + (k - d)] * Dinv[r];
    A[i] = (_Float16)v;
  }
}

__global__ void k_concat_ms_f16(const float* __restrict__ XS, int ps,
                                const float* __restrict__ XT, int pt,
                                const int* __restrict__ incp, int NS,
                                _Float16* __restrict__ A, int Kp, int d, long total) {
  for (long i = gidx(); i < total; i += gstr()) {
    long e = i / Kp; int k = (int)(i % Kp);
    float v;
    if (k < d) v = XS[e * (long)ps + k];
    else { int c = k - d; v = XT[(long)incp[NS + e] * pt + c] - XT[(long)incp[e] * pt + c]; }
    A[i] = (_Float16)v;
  }
}

__global__ void k_count_inc(const int* __restrict__ idx, long n, float* __restrict__ D) {
  for (long i = gidx(); i < n; i += gstr()) atomicAdd(&D[idx[i]], 1.f);
}
__global__ void k_invD(float* __restrict__ D, long n) {
  for (long i = gidx(); i < n; i += gstr()) D[i] = 1.f / (D[i] + 1e-6f);
}

__global__ void k_att_t(const float* __restrict__ XT, int pt, int coff,
                        const float* __restrict__ MT, const float* __restrict__ Dinv,
                        const float* __restrict__ W, const float* __restrict__ b,
                        float* __restrict__ att, int NT) {
  for (long n = gidx(); n < NT; n += gstr()) {
    float v = b[0]; float di = Dinv[n];
    for (int c = 0; c < 64; ++c) {
      v += XT[n * (long)pt + coff + c] * W[c];
      v += MT[n * 64 + c] * di * W[64 + c];
    }
    att[n] = 1.f / (1.f + expf(-v));
  }
}

__global__ void k_att_s(const float* __restrict__ XS, int ps, int coffs,
                        const float* __restrict__ XT, int pt, int cofft,
                        const int* __restrict__ incp, int NS,
                        const float* __restrict__ W, const float* __restrict__ b,
                        float* __restrict__ att) {
  for (long e = gidx(); e < NS; e += gstr()) {
    float v = b[0];
    long s = incp[e], d2 = incp[NS + e];
    for (int c = 0; c < 64; ++c) {
      v += XS[e * (long)ps + coffs + c] * W[c];
      v += (XT[d2 * pt + cofft + c] - XT[s * pt + cofft + c]) * W[64 + c];
    }
    att[e] = 1.f / (1.f + expf(-v));
  }
}

__global__ void k_pool_cnt(const int* __restrict__ pos, int N, int nseg, float* __restrict__ cnt) {
  for (long n = gidx(); n < N; n += gstr()) {
    int s = pos[n]; if (s < nseg) atomicAdd(&cnt[s], 1.f);
  }
}
__global__ void k_pool_sum(const float* __restrict__ X, int pin, int width,
                           const float* __restrict__ att, const int* __restrict__ pos,
                           int nseg, float* __restrict__ OUT, int pout, long total) {
  for (long i = gidx(); i < total; i += gstr()) {
    long r = i / width; int c = (int)(i % width);
    int s = pos[r];
    if (s < nseg) atomicAdd(&OUT[(long)s * pout + c], X[r * (long)pin + c] * att[r]);
  }
}
__global__ void k_pool_div(float* __restrict__ OUT, int pout, int width,
                           const float* __restrict__ cnt, long total) {
  for (long i = gidx(); i < total; i += gstr()) {
    long s = i / width; int c = (int)(i % width);
    OUT[s * (long)pout + c] /= fmaxf(cnt[s], 1.f);
  }
}

__global__ void k_dot_rows(const float* __restrict__ X, int pitch, int coff, int F,
                           const float* __restrict__ W, const float* __restrict__ b,
                           float* __restrict__ out, int N) {
  for (long n = gidx(); n < N; n += gstr()) {
    float v = b[0];
    for (int c = 0; c < F; ++c) v += X[n * (long)pitch + coff + c] * W[c];
    out[n] = v;
  }
}

__global__ void k_final(const float* __restrict__ xsr, const float* __restrict__ xtr,
                        const float* __restrict__ Wo, const float* __restrict__ bo,
                        float* __restrict__ out) {
  __shared__ float red[256];
  int b = blockIdx.x, t = threadIdx.x;
  float acc = 0.f;
  for (int i = t; i < 1089; i += 256) acc += xsr[b * 1089 + i] * Wo[i];
  for (int i = t; i < 1000; i += 256) acc += xtr[b * 1000 + i] * Wo[1089 + i];
  red[t] = acc; __syncthreads();
  for (int s = 128; s > 0; s >>= 1) { if (t < s) red[t] += red[t + s]; __syncthreads(); }
  if (t == 0) out[b] = red[0] + bo[0];
}

// ---------------- host helpers ----------------
static inline int gs(long tot) { long b = (tot + 255) / 256; if (b < 1) b = 1; if (b > 1048576) b = 1048576; return (int)b; }
static inline int pad32(int x) { return (x + 31) & ~31; }

struct Graph { const int* ei; const float* ew; int E; };

static void gemm(hipStream_t st, const _Float16* A, int Kp, const _Float16* Bt,
                 const float* bias, float* C, int ldc, int M, int Nact, int acc, int leaky) {
  int Ntiles = (Nact + 15) / 16;
  int Mtiles = M / 16;
  long waves = (long)Mtiles * Ntiles;
  int blocks = (int)((waves + 7) / 8);
  k_wmma_gemm<<<blocks, 256, 0, st>>>(A, Bt, bias, C, Kp, Kp, ldc,
                                      Mtiles, Ntiles, Nact, Kp, acc, leaky);
}

static void wconv(hipStream_t st, const float* W, int K, int N, _Float16* dst, int Kp, int Np) {
  long tot = (long)Np * Kp;
  k_wT_f16<<<gs(tot), 256, 0, st>>>(W, K, N, dst, Kp, Np, tot);
}

// Hodge-Laguerre conv (K=2) + BN + LeakyReLU, writing into OUT column slice
static void conv_block(hipStream_t st, const float* X, int px, int F, int M, Graph g,
                       const _Float16* W0, const _Float16* W1, const float* bias,
                       const float* gam, const float* bet,
                       float* OUT, int pout, int coff, int Fout,
                       float* XN, float* TMP, _Float16* A16, float* bns, float* bnq) {
  int Kp = pad32(F);
  long tc = (long)M * Kp;
  k_tof16<<<gs(tc), 256, 0, st>>>(X, px, 0, F, A16, Kp, tc);
  gemm(st, A16, Kp, W0, bias, XN, Fout, M, Fout, 0, 0);
  hipMemsetAsync(TMP, 0, (size_t)M * F * sizeof(float), st);
  long te = (long)g.E * F;
  k_spmm<<<gs(te), 256, 0, st>>>(g.ei, g.ew, g.E, X, px, F, TMP, te);
  long tm = (long)M * F;
  k_xmLx<<<gs(tm), 256, 0, st>>>(X, px, F, TMP, tm);
  k_tof16<<<gs(tc), 256, 0, st>>>(TMP, F, 0, F, A16, Kp, tc);
  gemm(st, A16, Kp, W1, nullptr, XN, Fout, M, Fout, 1, 0);
  hipMemsetAsync(bns, 0, Fout * sizeof(float), st);
  hipMemsetAsync(bnq, 0, Fout * sizeof(float), st);
  k_colstats<<<(M + 1023) / 1024, Fout, 0, st>>>(XN, M, Fout, 1024, bns, bnq);
  long tb = (long)M * Fout;
  k_bn_leaky<<<gs(tb), 256, 0, st>>>(XN, M, Fout, bns, bnq, gam, bet, OUT, pout, coff, tb);
}

// node<->edge interaction producing OT [NT,f], OS [NS,f]
static void neint(hipStream_t st, const float* XT0b, int pt, const float* XS0b, int ps,
                  int d, int NT, int NS, const int* incp, const float* Dinv,
                  const _Float16* Wt16, const float* bt, const _Float16* Ws16, const float* bs,
                  int f, float* OT, float* OS, float* MT, _Float16* A16) {
  hipMemsetAsync(MT, 0, (size_t)NT * d * sizeof(float), st);
  long te = (long)NS * d;
  k_scatter_mt<<<gs(te), 256, 0, st>>>(incp, NS, XS0b, ps, 0, d, MT, te);
  long tt = (long)NT * 2 * d;
  k_concat_mt_f16<<<gs(tt), 256, 0, st>>>(XT0b, pt, MT, d, Dinv, A16, 2 * d, tt);
  gemm(st, A16, 2 * d, Wt16, bt, OT, f, NT, f, 0, 0);
  long ts = (long)NS * 2 * d;
  k_concat_ms_f16<<<gs(ts), 256, 0, st>>>(XS0b, ps, XT0b, pt, incp, NS, A16, 2 * d, d, ts);
  gemm(st, A16, 2 * d, Ws16, bs, OS, f, NS, f, 0, 0);
}

// ---------------- entry ----------------
extern "C" void kernel_launch(void* const* d_in, const int* in_sizes, int n_in,
                              void* d_out, int out_size, void* d_ws, size_t ws_size,
                              hipStream_t stream) {
  hipStream_t st = stream;
  const float* x_t  = (const float*)d_in[0];
  const float* x_s  = (const float*)d_in[1];
  const float* ew_t = (const float*)d_in[2];
  const float* ew_s = (const float*)d_in[3];
  const float* ew_t2 = (const float*)d_in[4];
  const float* ew_s2 = (const float*)d_in[5];
  const float* Pf[92];
  for (int i = 0; i < 92; ++i) Pf[i] = (const float*)d_in[6 + i];
  const int* ei_t  = (const int*)d_in[98];
  const int* ei_s  = (const int*)d_in[99];
  const int* inc   = (const int*)d_in[100];
  const int* pos_t = (const int*)d_in[101];
  const int* pos_s = (const int*)d_in[102];
  const int* ei_t2 = (const int*)d_in[103];
  const int* ei_s2 = (const int*)d_in[104];
  const int* inc2  = (const int*)d_in[105];

  // bump allocator over workspace
  char* wp = (char*)d_ws;
  auto af = [&](size_t n) -> float* { float* r = (float*)wp; wp += ((n * 4 + 255) & ~(size_t)255); return r; };
  auto ah = [&](size_t n) -> _Float16* { _Float16* r = (_Float16*)wp; wp += ((n * 2 + 255) & ~(size_t)255); return r; };

  float* XT_EMB = af((size_t)NT1 * FT);
  float* XT0  = af((size_t)NT1 * 192);
  float* XS0  = af((size_t)NS1 * 192);
  float* XT0P = af((size_t)NT2 * 960);
  float* XS0P = af((size_t)NS2 * 960);
  float* OT  = af((size_t)NT1 * 64);
  float* OS  = af((size_t)NS1 * 64);
  float* XTN = af((size_t)NT1 * 64);
  float* XSN = af((size_t)NS1 * 64);
  float* TMP = af((size_t)NS1 * 64);
  float* MT  = af((size_t)NT1 * 128);
  _Float16* A16 = ah((size_t)NS1 * 256);
  float* DINV1 = af(NT1);
  float* DINV2 = af(NT2);
  float* ATT_T = af(NT1);
  float* ATT_S = af(NS1);
  float* CNT_T = af(NT2);
  float* CNT_S = af(NS2);
  float* BNS = af(256);
  float* BNQ = af(256);
  float* XTR = af(NT2);
  float* XSR = af(NS2);

  // ---- weights -> transposed f16 ----
  _Float16* W_emb = ah(80 * 96);
  wconv(st, Pf[P_emb_W], 74, 74, W_emb, 96, 80);
  _Float16* W_it0 = ah(64 * 96); _Float16* W_it1 = ah(64 * 96);
  wconv(st, Pf[P_init_t_W],           74, 64, W_it0, 96, 64);
  wconv(st, Pf[P_init_t_W] + 74 * 64, 74, 64, W_it1, 96, 64);
  _Float16* W_is0 = ah(64 * 32); _Float16* W_is1 = ah(64 * 32);
  wconv(st, Pf[P_init_s_W],           11, 64, W_is0, 32, 64);
  wconv(st, Pf[P_init_s_W] + 11 * 64, 11, 64, W_is1, 32, 64);

  _Float16 *W_ni_t[NLAY], *W_ni_s[NLAY], *W_cv_t0[NLAY], *W_cv_t1[NLAY], *W_cv_s0[NLAY], *W_cv_s1[NLAY];
  for (int l = 0; l < NLAY; ++l) {
    int f = Fl[l], Kd = 2 * Dl[l];
    int cb = P_cv_base + l * 8, nb = P_ni_base + l * 4;
    W_ni_t[l] = ah((size_t)f * Kd); wconv(st, Pf[nb + 1], Kd, f, W_ni_t[l], Kd, f);
    W_ni_s[l] = ah((size_t)f * Kd); wconv(st, Pf[nb + 0], Kd, f, W_ni_s[l], Kd, f);
    W_cv_t0[l] = ah((size_t)f * f); wconv(st, Pf[cb + 4],         f, f, W_cv_t0[l], f, f);
    W_cv_t1[l] = ah((size_t)f * f); wconv(st, Pf[cb + 4] + f * f, f, f, W_cv_t1[l], f, f);
    W_cv_s0[l] = ah((size_t)f * f); wconv(st, Pf[cb + 0],         f, f, W_cv_s0[l], f, f);
    W_cv_s1[l] = ah((size_t)f * f); wconv(st, Pf[cb + 0] + f * f, f, f, W_cv_s1[l], f, f);
  }

  // ---- degree / Dinv for level 1 ----
  hipMemsetAsync(DINV1, 0, NT1 * sizeof(float), st);
  k_count_inc<<<gs(2L * NS1), 256, 0, st>>>(inc, 2L * NS1, DINV1);
  k_invD<<<gs(NT1), 256, 0, st>>>(DINV1, NT1);

  // ---- embedding: xt = leaky(x_t @ emb_W + emb_b) ----
  {
    long tc = (long)NT1 * 96;
    k_tof16<<<gs(tc), 256, 0, st>>>(x_t, FT, 0, FT, A16, 96, tc);
    gemm(st, A16, 96, W_emb, Pf[P_emb_b], XT_EMB, FT, NT1, FT, 0, 1);
  }

  // ---- init HL convs ----
  conv_block(st, XT_EMB, FT, FT, NT1, Graph{ei_t, ew_t, MT1}, W_it0, W_it1,
             Pf[P_init_t_b], Pf[P_init_t_g], Pf[P_init_t_be],
             XT0, 192, 0, 64, XTN, TMP, A16, BNS, BNQ);
  conv_block(st, x_s, FS, FS, NS1, Graph{ei_s, ew_s, MS1}, W_is0, W_is1,
             Pf[P_init_s_b], Pf[P_init_s_g], Pf[P_init_s_be],
             XS0, 192, 0, 64, XSN, TMP, A16, BNS, BNQ);

  // ---- layer loop ----
  for (int l = 0; l < NLAY; ++l) {
    int f = Fl[l], d = Dl[l];
    bool lev1 = (l >= 2);
    int NT = lev1 ? NT2 : NT1, NS = lev1 ? NS2 : NS1;
    float* XT0b = lev1 ? XT0P : XT0; int pt = lev1 ? 960 : 192;
    float* XS0b = lev1 ? XS0P : XS0; int ps = pt;
    const int* incp = lev1 ? inc2 : inc;
    const float* Dinv = lev1 ? DINV2 : DINV1;
    Graph gt = lev1 ? Graph{ei_t2, ew_t2, MT2} : Graph{ei_t, ew_t, MT1};
    Graph gsg = lev1 ? Graph{ei_s2, ew_s2, MS2} : Graph{ei_s, ew_s, MS1};
    int nb = P_ni_base + l * 4, cb = P_cv_base + l * 8;

    neint(st, XT0b, pt, XS0b, ps, d, NT, NS, incp, Dinv,
          W_ni_t[l], Pf[nb + 3], W_ni_s[l], Pf[nb + 2], f, OT, OS, MT, A16);

    conv_block(st, OT, f, f, NT, gt, W_cv_t0[l], W_cv_t1[l],
               Pf[cb + 5], Pf[cb + 7], Pf[cb + 6],
               XT0b, pt, d, f, XTN, TMP, A16, BNS, BNQ);
    conv_block(st, OS, f, f, NS, gsg, W_cv_s0[l], W_cv_s1[l],
               Pf[cb + 1], Pf[cb + 3], Pf[cb + 2],
               XS0b, ps, d, f, XSN, TMP, A16, BNS, BNQ);

    if (l == 1) {  // attention + pooling after i==0
      // attention over last-layer outputs (cols 128..191)
      hipMemsetAsync(MT, 0, (size_t)NT1 * 64 * sizeof(float), st);
      long te = (long)NS1 * 64;
      k_scatter_mt<<<gs(te), 256, 0, st>>>(inc, NS1, XS0, 192, 128, 64, MT, te);
      k_att_t<<<gs(NT1), 256, 0, st>>>(XT0, 192, 128, MT, DINV1,
                                       Pf[P_att_Wt], Pf[P_att_bt], ATT_T, NT1);
      k_att_s<<<gs(NS1), 256, 0, st>>>(XS0, 192, 128, XT0, 192, 128, inc, NS1,
                                       Pf[P_att_Ws], Pf[P_att_bs], ATT_S);
      // segment-mean pooling of attention-scaled features
      hipMemsetAsync(XT0P, 0, (size_t)NT2 * 960 * sizeof(float), st);
      hipMemsetAsync(XS0P, 0, (size_t)NS2 * 960 * sizeof(float), st);
      hipMemsetAsync(CNT_T, 0, NT2 * sizeof(float), st);
      hipMemsetAsync(CNT_S, 0, NS2 * sizeof(float), st);
      k_pool_cnt<<<gs(NT1), 256, 0, st>>>(pos_t, NT1, NT2, CNT_T);
      k_pool_cnt<<<gs(NS1), 256, 0, st>>>(pos_s, NS1, NS2, CNT_S);
      long tt = (long)NT1 * 192;
      k_pool_sum<<<gs(tt), 256, 0, st>>>(XT0, 192, 192, ATT_T, pos_t, NT2, XT0P, 960, tt);
      long ts = (long)NS1 * 192;
      k_pool_sum<<<gs(ts), 256, 0, st>>>(XS0, 192, 192, ATT_S, pos_s, NS2, XS0P, 960, ts);
      long dt = (long)NT2 * 192;
      k_pool_div<<<gs(dt), 256, 0, st>>>(XT0P, 960, 192, CNT_T, dt);
      long dsn = (long)NS2 * 192;
      k_pool_div<<<gs(dsn), 256, 0, st>>>(XS0P, 960, 192, CNT_S, dsn);
      // Dinv for level 2
      hipMemsetAsync(DINV2, 0, NT2 * sizeof(float), st);
      k_count_inc<<<gs(2L * NS2), 256, 0, st>>>(inc2, 2L * NS2, DINV2);
      k_invD<<<gs(NT2), 256, 0, st>>>(DINV2, NT2);
    }
  }

  // ---- readout (K=1 lag) + final linear ----
  k_dot_rows<<<gs(NT2), 256, 0, st>>>(XT0P, 960, 704, 256, Pf[P_ro_tW], Pf[P_ro_tb], XTR, NT2);
  k_dot_rows<<<gs(NS2), 256, 0, st>>>(XS0P, 960, 704, 256, Pf[P_ro_sW], Pf[P_ro_sb], XSR, NS2);
  k_final<<<16, 256, 0, st>>>(XSR, XTR, Pf[P_out_W], Pf[P_out_b], (float*)d_out);

  (void)in_sizes; (void)n_in; (void)out_size; (void)ws_size;
}
